// GCN_21990232555546
// MI455X (gfx1250) — matrix operand order repaired and checked by reading, not compiled, not run on hardware
//
#include <hip/hip_runtime.h>
#include <hip/hip_bf16.h>

typedef __attribute__((ext_vector_type(16))) __bf16 v16bf;
typedef __attribute__((ext_vector_type(8)))  float  v8f;

#define TPB 256

// ---------------- utility ----------------
__global__ void fill0_kernel(float* __restrict__ p, long long n) {
  long long i = (long long)blockIdx.x * blockDim.x + threadIdx.x;
  if (i < n) p[i] = 0.0f;
}

// ---------------- normalization ----------------
__global__ void deg_kernel(const long long* __restrict__ src, const float* __restrict__ w,
                           float* __restrict__ deg, int nE) {
  int e = blockIdx.x * blockDim.x + threadIdx.x;
  if (e < nE) atomicAdd(&deg[(int)src[e]], w[e]);
}

__global__ void dinv_kernel(float* __restrict__ deg, int n) {
  int i = blockIdx.x * blockDim.x + threadIdx.x;
  if (i < n) {
    float d = deg[i];
    deg[i] = (d > 0.0f) ? rsqrtf(fmaxf(d, 1e-30f)) : 0.0f;
  }
}

__global__ void normw_kernel(const long long* __restrict__ src, const long long* __restrict__ dst,
                             const float* __restrict__ w, const float* __restrict__ dinv,
                             float* __restrict__ normw, int nE) {
  int e = blockIdx.x * blockDim.x + threadIdx.x;
  if (e < nE) normw[e] = -dinv[(int)src[e]] * w[e] * dinv[(int)dst[e]];
}

// ---------------- sparse propagation: y[dst] += w_e * x[src] ----------------
__global__ void prop_kernel(const float* __restrict__ x, const long long* __restrict__ src,
                            const long long* __restrict__ dst, const float* __restrict__ w,
                            float* __restrict__ y, int F, int nE, int lf4) {
  long long t = (long long)blockIdx.x * blockDim.x + threadIdx.x;
  if (t >= ((long long)nE << lf4)) return;
  int e  = (int)(t >> lf4);
  int f4 = (int)(t & ((1 << lf4) - 1));
  float we = w[e];
  int s = (int)src[e], d = (int)dst[e];
  float4 v = *(const float4*)(x + (long long)s * F + 4 * f4);
  float* yp = y + (long long)d * F + 4 * f4;
  atomicAdd(yp + 0, we * v.x);
  atomicAdd(yp + 1, we * v.y);
  atomicAdd(yp + 2, we * v.z);
  atomicAdd(yp + 3, we * v.w);
}

// ---------------- weight pre-pack into WMMA B-fragment order (bf16) ----------------
// Layout: [term][kbi][tn][lane(32)][16 values], each lane's 16 values contiguous
// (32 bytes) so the GEMM loads a full B fragment with one v16bf load.
// Lane's values: column n = tn*16 + (lane&15), K = kbi*32 + (lane>>4)*8 + (t&7) + (t>>3)*16, t=0..15.
__global__ void packW_kernel(const float* __restrict__ W, __bf16* __restrict__ out,
                             int Fin, int Fout) {
  int tid = blockIdx.x * blockDim.x + threadIdx.x;
  int total = 3 * Fin * Fout;
  if (tid >= total) return;
  const int tilesN = Fout >> 4;
  const int nKb = Fin >> 5;
  int t    = tid & 15;
  int lane = (tid >> 4) & 31;
  int rest = tid >> 9;                   // (term*nKb + kbi)*tilesN + tn
  int tn   = rest % tilesN;  rest /= tilesN;
  int kbi  = rest % nKb;     rest /= nKb;
  int term = rest;
  int k = (kbi << 5) + (lane >> 4) * 8 + (t & 7) + (t >> 3) * 16;
  int n = (tn << 4) + (lane & 15);
  out[tid] = (__bf16)W[((long long)term * Fin + k) * Fout + n];
}

// ---------------- fused Chebyshev GEMM ----------------
// out = relu( T0@W[0] + T1@W[1] + (2*T2raw - T0)@W[2] + bias )
// One 16x16 tile per wave32; FIN/FOUT compile-time so all loops fully unroll.
template <int FIN, int FOUT>
__global__ void __launch_bounds__(TPB) cheb_gemm_kernel(
    const float* __restrict__ T0, const float* __restrict__ T1,
    const float* __restrict__ T2, const __bf16* __restrict__ Wp,
    const float* __restrict__ bias, float* __restrict__ out, int M)
{
  constexpr int tilesN = FOUT / 16;
  constexpr int nKb    = FIN / 32;
  const int wave = threadIdx.x >> 5;
  const int lane = threadIdx.x & 31;
  const int totalTiles = (M >> 4) * tilesN;
  const int tile = blockIdx.x * (TPB / 32) + wave;
  if (tile >= totalTiles) return;          // wave-uniform: EXEC stays all-ones
  const int tm = tile / tilesN;
  const int tn = tile % tilesN;
  const int hf  = lane >> 4;
  const int l16 = lane & 15;
  const int ncol = tn * 16 + l16;
  const long long rowbase = (long long)(tm * 16 + l16) * FIN;

  v8f acc = {};
#pragma unroll
  for (int term = 0; term < 3; ++term) {
    const float* Aptr = (term == 0) ? T0 : (term == 1) ? T1 : T2;
#pragma unroll
    for (int kbi = 0; kbi < nKb; ++kbi) {
      const int kb = kbi * 32;
      // A fragment: two contiguous 8-float chunks per lane -> 4x float4 loads.
      const float* ap = Aptr + rowbase + kb + hf * 8;
      float4 a0 = *(const float4*)(ap);
      float4 a1 = *(const float4*)(ap + 4);
      float4 a2 = *(const float4*)(ap + 16);
      float4 a3 = *(const float4*)(ap + 20);
      if (term == 2) {                     // T2 = 2*prop(T1) - T0 on the fly
        const float* qp = T0 + rowbase + kb + hf * 8;
        float4 q0 = *(const float4*)(qp);
        float4 q1 = *(const float4*)(qp + 4);
        float4 q2 = *(const float4*)(qp + 16);
        float4 q3 = *(const float4*)(qp + 20);
        a0.x = 2.0f*a0.x - q0.x; a0.y = 2.0f*a0.y - q0.y; a0.z = 2.0f*a0.z - q0.z; a0.w = 2.0f*a0.w - q0.w;
        a1.x = 2.0f*a1.x - q1.x; a1.y = 2.0f*a1.y - q1.y; a1.z = 2.0f*a1.z - q1.z; a1.w = 2.0f*a1.w - q1.w;
        a2.x = 2.0f*a2.x - q2.x; a2.y = 2.0f*a2.y - q2.y; a2.z = 2.0f*a2.z - q2.z; a2.w = 2.0f*a2.w - q2.w;
        a3.x = 2.0f*a3.x - q3.x; a3.y = 2.0f*a3.y - q3.y; a3.z = 2.0f*a3.z - q3.z; a3.w = 2.0f*a3.w - q3.w;
      }
      v16bf a;
      a[0]  = (__bf16)a0.x; a[1]  = (__bf16)a0.y; a[2]  = (__bf16)a0.z; a[3]  = (__bf16)a0.w;
      a[4]  = (__bf16)a1.x; a[5]  = (__bf16)a1.y; a[6]  = (__bf16)a1.z; a[7]  = (__bf16)a1.w;
      a[8]  = (__bf16)a2.x; a[9]  = (__bf16)a2.y; a[10] = (__bf16)a2.z; a[11] = (__bf16)a2.w;
      a[12] = (__bf16)a3.x; a[13] = (__bf16)a3.y; a[14] = (__bf16)a3.z; a[15] = (__bf16)a3.w;
      // B fragment: one contiguous 32B chunk per lane from the packed weights.
      v16bf b = *(const v16bf*)(Wp +
          ((((long long)term * nKb + kbi) * tilesN + tn) * 32 + lane) * 16);
      acc = __builtin_amdgcn_wmma_f32_16x16x32_bf16(
          false, a, false, b, (short)0, acc, false, false);
    }
  }

  const float bv = bias[ncol];
#pragma unroll
  for (int i = 0; i < 8; ++i) {            // C/D: VGPR i -> row i + 8*hf
    const int m = tm * 16 + hf * 8 + i;
    out[(long long)m * FOUT + ncol] = fmaxf(acc[i] + bv, 0.0f);
  }
}

// ---------------- global mean pool ----------------
__global__ void pool_kernel(const float* __restrict__ h, const long long* __restrict__ batch,
                            float* __restrict__ pooled, float* __restrict__ counts, int nN) {
  long long t = (long long)blockIdx.x * blockDim.x + threadIdx.x;
  if (t >= (long long)nN * 128) return;
  int n = (int)(t >> 7);
  int f = (int)(t & 127);
  int b = (int)batch[n];
  atomicAdd(&pooled[b * 128 + f], h[(long long)n * 128 + f]);
  if (f == 0) atomicAdd(&counts[b], 1.0f);
}

__global__ void final_linear_kernel(const float* __restrict__ pooled, const float* __restrict__ counts,
                                    const float* __restrict__ linW, const float* __restrict__ linb,
                                    float* __restrict__ out) {
  int idx = blockIdx.x * blockDim.x + threadIdx.x;
  if (idx >= 64 * 10) return;
  int g = idx / 10, o = idx % 10;
  float inv = 1.0f / fmaxf(counts[g], 1.0f);
  float s = 0.0f;
  for (int k = 0; k < 128; ++k)
    s += pooled[g * 128 + k] * inv * linW[k * 10 + o];
  out[idx] = s + linb[o];
}

// ---------------- host-side orchestration ----------------
static inline unsigned gridFor(long long n) { return (unsigned)((n + TPB - 1) / TPB); }

extern "C" void kernel_launch(void* const* d_in, const int* in_sizes, int n_in,
                              void* d_out, int out_size, void* d_ws, size_t ws_size,
                              hipStream_t stream) {
  const float*     x    = (const float*)d_in[0];     // (nN, 128)
  const long long* ei   = (const long long*)d_in[1]; // (2, nE) int64
  const float*     ew   = (const float*)d_in[2];     // (nE,)
  const long long* batch= (const long long*)d_in[3]; // (nN,) int64
  const float*     W1   = (const float*)d_in[4];     // (3,128,64)
  const float*     b1   = (const float*)d_in[5];
  const float*     W2   = (const float*)d_in[6];     // (3,64,64)
  const float*     b2   = (const float*)d_in[7];
  const float*     W3   = (const float*)d_in[8];     // (3,64,128)
  const float*     b3   = (const float*)d_in[9];
  const float*     linW = (const float*)d_in[10];    // (128,10)
  const float*     linb = (const float*)d_in[11];
  float* out = (float*)d_out;                        // (64,10)

  const int nN = in_sizes[0] / 128;
  const int nE = in_sizes[1] / 2;
  const long long featCap = (long long)nN * 128;

  const long long* src = ei;
  const long long* dst = ei + nE;

  // workspace layout (f32 elements, then bf16 packed weights)
  float* ws     = (float*)d_ws;
  float* deg    = ws;                      // nN
  float* normw  = deg    + nN;             // nE
  float* t1     = normw  + nE;             // nN*128
  float* t2     = t1     + featCap;        // nN*128
  float* fA     = t2     + featCap;        // nN*128
  float* fB     = fA     + featCap;        // nN*128
  float* pooled = fB     + featCap;        // 64*128
  float* counts = pooled + 64 * 128;       // 64
  uintptr_t pwAddr = ((uintptr_t)(counts + 64) + 31) & ~(uintptr_t)31;
  __bf16* pw1 = (__bf16*)pwAddr;           // 3*128*64
  __bf16* pw2 = pw1 + 3 * 128 * 64;        // 3*64*64
  __bf16* pw3 = pw2 + 3 * 64 * 64;         // 3*64*128

  // ---- pack weights into WMMA fragment order (cheap, every call) ----
  packW_kernel<<<gridFor(3 * 128 * 64), TPB, 0, stream>>>(W1, pw1, 128, 64);
  packW_kernel<<<gridFor(3 * 64 * 64),  TPB, 0, stream>>>(W2, pw2, 64, 64);
  packW_kernel<<<gridFor(3 * 64 * 128), TPB, 0, stream>>>(W3, pw3, 64, 128);

  // ---- symmetric normalization weights ----
  fill0_kernel<<<gridFor(nN), TPB, 0, stream>>>(deg, nN);
  deg_kernel<<<gridFor(nE), TPB, 0, stream>>>(src, ew, deg, nE);
  dinv_kernel<<<gridFor(nN), TPB, 0, stream>>>(deg, nN);
  normw_kernel<<<gridFor(nE), TPB, 0, stream>>>(src, dst, ew, deg, normw, nE);

  // ---- layer 1: 128 -> 64 ----
  {
    const int Fin = 128, Fout = 64, lf4 = 5;
    fill0_kernel<<<gridFor((long long)nN * Fin), TPB, 0, stream>>>(t1, (long long)nN * Fin);
    prop_kernel<<<gridFor((long long)nE << lf4), TPB, 0, stream>>>(x, src, dst, normw, t1, Fin, nE, lf4);
    fill0_kernel<<<gridFor((long long)nN * Fin), TPB, 0, stream>>>(t2, (long long)nN * Fin);
    prop_kernel<<<gridFor((long long)nE << lf4), TPB, 0, stream>>>(t1, src, dst, normw, t2, Fin, nE, lf4);
    const int tiles = (nN / 16) * (Fout / 16);
    cheb_gemm_kernel<128, 64><<<(tiles + TPB/32 - 1) / (TPB/32), TPB, 0, stream>>>(
        x, t1, t2, pw1, b1, fA, nN);
  }

  // ---- layer 2: 64 -> 64 ----
  {
    const int Fin = 64, Fout = 64, lf4 = 4;
    fill0_kernel<<<gridFor((long long)nN * Fin), TPB, 0, stream>>>(t1, (long long)nN * Fin);
    prop_kernel<<<gridFor((long long)nE << lf4), TPB, 0, stream>>>(fA, src, dst, normw, t1, Fin, nE, lf4);
    fill0_kernel<<<gridFor((long long)nN * Fin), TPB, 0, stream>>>(t2, (long long)nN * Fin);
    prop_kernel<<<gridFor((long long)nE << lf4), TPB, 0, stream>>>(t1, src, dst, normw, t2, Fin, nE, lf4);
    const int tiles = (nN / 16) * (Fout / 16);
    cheb_gemm_kernel<64, 64><<<(tiles + TPB/32 - 1) / (TPB/32), TPB, 0, stream>>>(
        fA, t1, t2, pw2, b2, fB, nN);
  }

  // ---- layer 3: 64 -> 128 ----
  {
    const int Fin = 64, Fout = 128, lf4 = 4;
    fill0_kernel<<<gridFor((long long)nN * Fin), TPB, 0, stream>>>(t1, (long long)nN * Fin);
    prop_kernel<<<gridFor((long long)nE << lf4), TPB, 0, stream>>>(fB, src, dst, normw, t1, Fin, nE, lf4);
    fill0_kernel<<<gridFor((long long)nN * Fin), TPB, 0, stream>>>(t2, (long long)nN * Fin);
    prop_kernel<<<gridFor((long long)nE << lf4), TPB, 0, stream>>>(t1, src, dst, normw, t2, Fin, nE, lf4);
    const int tiles = (nN / 16) * (Fout / 16);
    cheb_gemm_kernel<64, 128><<<(tiles + TPB/32 - 1) / (TPB/32), TPB, 0, stream>>>(
        fB, t1, t2, pw3, b3, fA, nN);
  }

  // ---- global mean pool + classifier ----
  fill0_kernel<<<gridFor(64 * 128 + 64), TPB, 0, stream>>>(pooled, 64 * 128 + 64);
  pool_kernel<<<gridFor((long long)nN * 128), TPB, 0, stream>>>(fA, batch, pooled, counts, nN);
  final_linear_kernel<<<gridFor(640), TPB, 0, stream>>>(pooled, counts, linW, linb, out);
}
